// BiLSTM_CRF_70385924046978
// MI455X (gfx1250) — matrix operand order repaired and checked by reading, not compiled, not run on hardware
//
#include <hip/hip_runtime.h>
#include <hip/hip_bf16.h>
#include <math.h>

// Problem constants (match reference)
#define Bn 64
#define Tn 256
#define Fn 256
#define Hn 512
#define Cn 32

typedef __attribute__((ext_vector_type(16))) __bf16 v16bf;
typedef __attribute__((ext_vector_type(8)))  float  v8f;
typedef __attribute__((ext_vector_type(4)))  unsigned int v4u;
typedef __attribute__((ext_vector_type(4)))  int v4i;
typedef unsigned short ushort_t;

union FragU { v4u q[2]; v16bf h; };

// ---- CDNA5 async global->LDS path (guarded; falls back to sync staging) ----
#if defined(__has_builtin)
#  if __has_builtin(__builtin_amdgcn_global_load_async_to_lds_b128) && \
      __has_builtin(__builtin_amdgcn_s_wait_asynccnt)
#    define HAVE_ASYNC 1
#  endif
#endif
#ifndef HAVE_ASYNC
#  define HAVE_ASYNC 0
#endif

#if HAVE_ASYNC
typedef __attribute__((address_space(1))) v4i* gv4i_p;  // global v4i*
typedef __attribute__((address_space(3))) v4i* lv4i_p;  // LDS v4i*
#endif

__device__ inline void async_cp16(const ushort_t* g, ushort_t* l) {
#if HAVE_ASYNC
  __builtin_amdgcn_global_load_async_to_lds_b128(
      (gv4i_p)(void*)g, (lv4i_p)(void*)l, 0, 0);
#else
  *(v4u*)l = *(const v4u*)g;
#endif
}

#if HAVE_ASYNC
#  define WAIT_ASYNC(n) __builtin_amdgcn_s_wait_asynccnt(n)
#else
#  define WAIT_ASYNC(n)
#endif

__device__ inline ushort_t f2bf(float f) {
  unsigned u = __float_as_uint(f);
  u = u + 0x7FFFu + ((u >> 16) & 1u);   // round-to-nearest-even
  return (ushort_t)(u >> 16);
}

__device__ inline float sigf(float x) { return 1.0f / (1.0f + __expf(-x)); }
__device__ inline float tanh_fast(float x) {
  x = fminf(fmaxf(x, -15.0f), 15.0f);
  float e = __expf(2.0f * x);
  return (e - 1.0f) / (e + 1.0f);
}

// A-matrix fragment (16x32 bf16, MxK) from global: lane L: M=L&15,
// K chunks [half*8,+8) in v0..3 and [16+half*8,+8) in v4..7 (ISA 7.12.2)
__device__ inline v16bf load_a_g(const ushort_t* p, int ld, int row0, int k0, int lane) {
  int r = lane & 15, half = lane >> 4;
  const ushort_t* base = p + (size_t)(row0 + r) * ld + k0 + half * 8;
  FragU f;
  f.q[0] = *(const v4u*)(base);
  f.q[1] = *(const v4u*)(base + 16);
  return f.h;
}

// B-matrix fragment (32x16 bf16, KxN) from row-major (N,K) weights:
// lane L: N=L&15, K chunks [half*8,+8) in v0..3 and [16+half*8,+8) in v4..7 (ISA 7.12.2 / 7.12.4)
__device__ inline v16bf load_b_g(const ushort_t* p, int ld, int col0, int k0, int lane) {
  int r = lane & 15, half = lane >> 4;
  const ushort_t* base = p + (size_t)(col0 + r) * ld + k0 + half * 8;
  FragU f;
  f.q[0] = *(const v4u*)(base);
  f.q[1] = *(const v4u*)(base + 16);
  return f.h;
}

// LDS swizzles: XOR 16B-chunk index with low row bits to spread banks
__device__ inline int shswz(int b, int j) {          // 512-element rows (h state)
  return b * Hn + ((((j >> 3) ^ (b & 7)) << 3) | (j & 7));
}
__device__ inline int kswz(int row, int col) {       // 32-element rows (GEMM slabs)
  return row * 32 + (((((col >> 3) ^ row) & 3) << 3) | (col & 7));
}

// ---------------------------------------------------------------- cvt f32->bf16
__global__ void cvt_bf16(const float* __restrict__ in, ushort_t* __restrict__ out,
                         long long n) {
  long long i = (long long)blockIdx.x * blockDim.x + threadIdx.x;
  if (i < n) out[i] = f2bf(in[i]);
}

// ---------------------------------------------------------------- direct GEMM
// Small-N path (emissions): one wave32 per 16x16 tile, operands straight to VGPRs.
__global__ __launch_bounds__(256) void gemm_bf16_wmma(
    const ushort_t* __restrict__ A, int lda,
    const ushort_t* __restrict__ W, int ldw,
    const float* __restrict__ bias,
    float* __restrict__ C, int ldc,
    int M, int N, int K) {
  int lane = threadIdx.x & 31;
  int wid  = blockIdx.x * (blockDim.x >> 5) + (threadIdx.x >> 5);
  int ntn  = N >> 4;
  int tm = wid / ntn, tn = wid % ntn;
  if (tm * 16 >= M) return;

  v8f acc;
#pragma unroll
  for (int v = 0; v < 8; ++v) acc[v] = 0.0f;

  for (int k0 = 0; k0 < K; k0 += 32) {
    v16bf a = load_a_g(A, lda, tm * 16, k0, lane);
    v16bf b = load_b_g(W, ldw, tn * 16, k0, lane);
    if (k0 + 32 < K) {  // global_prefetch_b8 of next K-slab
      __builtin_prefetch(A + (size_t)(tm * 16 + (lane & 15)) * lda + k0 + 32, 0, 1);
      __builtin_prefetch(W + (size_t)(tn * 16 + (lane & 15)) * ldw + k0 + 32, 0, 1);
    }
    acc = __builtin_amdgcn_wmma_f32_16x16x32_bf16(false, a, false, b,
                                                  (short)0, acc, false, false);
  }

  int r = lane & 15, half = lane >> 4;
  float bv = bias ? bias[tn * 16 + r] : 0.0f;
#pragma unroll
  for (int v = 0; v < 8; ++v)
    C[(size_t)(tm * 16 + v + 8 * half) * ldc + tn * 16 + r] = acc[v] + bv;
}

// ---------------------------------------------------------------- LDS-staged GEMM
// Big projections: block tile 128x64, K-slabs of 32, operands staged to LDS via
// async global->LDS copies, double-buffered on ASYNCcnt. 8 waves, each owning a
// 32x32 quadrant (2x2 WMMA tiles) -> every staged element reused from LDS.
#define BM 128
#define BN 64
#define BK 32

__device__ inline void stage_slab(const ushort_t* A, int lda, int m0,
                                  const ushort_t* W, int ldw, int n0,
                                  int k0, ushort_t* As, ushort_t* Bs, int tid) {
#pragma unroll
  for (int i = 0; i < 2; ++i) {       // A: 128 rows x 4 chunks = 512, 2/thread
    int c = tid + i * 256;
    int row = c >> 2, ch = c & 3;
    async_cp16(A + (size_t)(m0 + row) * lda + k0 + ch * 8, &As[kswz(row, ch * 8)]);
  }
  {                                   // B: 64 rows x 4 chunks = 256, 1/thread
    int row = tid >> 2, ch = tid & 3;
    async_cp16(W + (size_t)(n0 + row) * ldw + k0 + ch * 8, &Bs[kswz(row, ch * 8)]);
  }
}

__global__ __launch_bounds__(256) void gemm_bf16_lds(
    const ushort_t* __restrict__ A, int lda,
    const ushort_t* __restrict__ W, int ldw,
    const float* __restrict__ bias,
    float* __restrict__ C, int ldc,
    int M, int N, int K) {
  __shared__ ushort_t As[2][BM * BK];   // 2 x 8KB
  __shared__ ushort_t Bs[2][BN * BK];   // 2 x 4KB
  const int tid = threadIdx.x, lane = tid & 31, w = tid >> 5;
  const int r = lane & 15, half = lane >> 4;
  const int nbn = N / BN;
  const int bm = blockIdx.x / nbn, bn = blockIdx.x % nbn;
  const int m0 = bm * BM, n0 = bn * BN;
  const int wm = w >> 1, wn = w & 1;

  v8f acc[2][2];
#pragma unroll
  for (int i = 0; i < 2; ++i)
#pragma unroll
    for (int j = 0; j < 2; ++j)
#pragma unroll
      for (int v = 0; v < 8; ++v) acc[i][j][v] = 0.0f;

  const int nk = K / BK;
  stage_slab(A, lda, m0, W, ldw, n0, 0, As[0], Bs[0], tid);

  for (int ks = 0; ks < nk; ++ks) {
    const int cur = ks & 1;
    if (ks + 1 < nk) {  // keep next slab's 3 per-thread copies in flight
      stage_slab(A, lda, m0, W, ldw, n0, (ks + 1) * BK, As[cur ^ 1], Bs[cur ^ 1], tid);
      WAIT_ASYNC(3);
    } else {
      WAIT_ASYNC(0);
    }
    __syncthreads();

    FragU a0, a1, b0, b1;
    const int ar0 = wm * 32 + r, ar1 = ar0 + 16;
    a0.q[0] = *(const v4u*)&As[cur][kswz(ar0, half * 8)];
    a0.q[1] = *(const v4u*)&As[cur][kswz(ar0, 16 + half * 8)];
    a1.q[0] = *(const v4u*)&As[cur][kswz(ar1, half * 8)];
    a1.q[1] = *(const v4u*)&As[cur][kswz(ar1, 16 + half * 8)];
    const int bc0 = wn * 32 + r, bc1 = bc0 + 16;
    b0.q[0] = *(const v4u*)&Bs[cur][kswz(bc0, half * 8)];
    b0.q[1] = *(const v4u*)&Bs[cur][kswz(bc0, 16 + half * 8)];
    b1.q[0] = *(const v4u*)&Bs[cur][kswz(bc1, half * 8)];
    b1.q[1] = *(const v4u*)&Bs[cur][kswz(bc1, 16 + half * 8)];

    acc[0][0] = __builtin_amdgcn_wmma_f32_16x16x32_bf16(false, a0.h, false, b0.h,
                                                        (short)0, acc[0][0], false, false);
    acc[0][1] = __builtin_amdgcn_wmma_f32_16x16x32_bf16(false, a0.h, false, b1.h,
                                                        (short)0, acc[0][1], false, false);
    acc[1][0] = __builtin_amdgcn_wmma_f32_16x16x32_bf16(false, a1.h, false, b0.h,
                                                        (short)0, acc[1][0], false, false);
    acc[1][1] = __builtin_amdgcn_wmma_f32_16x16x32_bf16(false, a1.h, false, b1.h,
                                                        (short)0, acc[1][1], false, false);
    __syncthreads();  // don't overwrite 'cur' buffers while others still read
  }

#pragma unroll
  for (int ti = 0; ti < 2; ++ti)
#pragma unroll
    for (int tj = 0; tj < 2; ++tj) {
      const int nbase = n0 + wn * 32 + tj * 16;
      const float bv = bias ? bias[nbase + r] : 0.0f;
#pragma unroll
      for (int v = 0; v < 8; ++v)
        C[(size_t)(m0 + wm * 32 + ti * 16 + v + 8 * half) * ldc + nbase + r] =
            acc[ti][tj][v] + bv;
    }
}

// ---------------------------------------------------------------- LSTM recurrence
// One persistent 1024-thread workgroup per direction (blockIdx.x = dir).
// h state: bf16 in LDS (64KB, swizzled). c state: VGPRs (wave owns fixed tiles).
// Wave w owns hidden tile jt=w for ALL four batch tiles, processed in two
// bt-pairs so every W_hh fragment loaded from L2 feeds two WMMAs.
__global__ __launch_bounds__(1024) void bilstm_rec(
    const float* __restrict__ preF, const float* __restrict__ preB,  // (B,T,4H)
    const ushort_t* __restrict__ WhhF, const ushort_t* __restrict__ WhhB, // (4H,H) bf16
    float* __restrict__ hs) {  // (B,T,2H) f32; fwd -> cols [0,H), bwd -> [H,2H)
  __shared__ ushort_t sh[Bn * Hn];  // 64 x 512 bf16 = 64KB

  const int dir = blockIdx.x;
  const float*    pre = dir ? preB : preF;
  const ushort_t* Whh = dir ? WhhB : WhhF;
  const int colOff = dir ? Hn : 0;

  const int tid = threadIdx.x, lane = tid & 31, w = tid >> 5;
  const int r = lane & 15, half = lane >> 4;
  const int jt = w;  // hidden tile owned by this wave (32 waves x 16 cols = 512)

  for (int i = tid; i < Bn * Hn; i += 1024) sh[i] = 0;  // h(-1) = 0

  float cst[4][8];
#pragma unroll
  for (int bt = 0; bt < 4; ++bt)
#pragma unroll
    for (int v = 0; v < 8; ++v) cst[bt][v] = 0.0f;

  __syncthreads();

  for (int t = 0; t < Tn; ++t) {
    const int tt = dir ? (Tn - 1 - t) : t;
    float hval[4][8];

#pragma unroll
    for (int pass = 0; pass < 2; ++pass) {   // bt pair {2*pass, 2*pass+1}
      v8f acc[2][4];
#pragma unroll
      for (int p = 0; p < 2; ++p) {          // init accumulators from pre (i,f,g,o)
        const int bt = pass * 2 + p;
#pragma unroll
        for (int g = 0; g < 4; ++g) {
          const int col = g * Hn + jt * 16 + r;
#pragma unroll
          for (int v = 0; v < 8; ++v) {
            const int b = bt * 16 + v + 8 * half;
            acc[p][g][v] = pre[((size_t)b * Tn + tt) * (4 * Hn) + col];
          }
        }
      }

      for (int k0 = 0; k0 < Hn; k0 += 32) {
        const int b0r = (pass * 2) * 16 + r;       // A-frag rows = batch, from LDS
        const int b1r = b0r + 16;
        FragU fa0, fa1;
        fa0.q[0] = *(const v4u*)&sh[shswz(b0r, k0 + half * 8)];
        fa0.q[1] = *(const v4u*)&sh[shswz(b0r, k0 + 16 + half * 8)];
        fa1.q[0] = *(const v4u*)&sh[shswz(b1r, k0 + half * 8)];
        fa1.q[1] = *(const v4u*)&sh[shswz(b1r, k0 + 16 + half * 8)];
#pragma unroll
        for (int g = 0; g < 4; ++g) {              // each B-frag feeds 2 WMMAs
          v16bf bf = load_b_g(Whh, Hn, g * Hn + jt * 16, k0, lane);
          acc[0][g] = __builtin_amdgcn_wmma_f32_16x16x32_bf16(
              false, fa0.h, false, bf, (short)0, acc[0][g], false, false);
          acc[1][g] = __builtin_amdgcn_wmma_f32_16x16x32_bf16(
              false, fa1.h, false, bf, (short)0, acc[1][g], false, false);
        }
      }

#pragma unroll
      for (int p = 0; p < 2; ++p) {          // gate nonlinearities (i,f,g,o)
        const int bt = pass * 2 + p;
#pragma unroll
        for (int v = 0; v < 8; ++v) {
          float iv = sigf(acc[p][0][v]);
          float fv = sigf(acc[p][1][v]);
          float gv = tanh_fast(acc[p][2][v]);
          float ov = sigf(acc[p][3][v]);
          float c  = fv * cst[bt][v] + iv * gv;
          cst[bt][v]  = c;
          hval[bt][v] = ov * tanh_fast(c);
        }
      }
    }

    __syncthreads();  // everyone done reading h(t-1) from LDS

    const int j = jt * 16 + r;
#pragma unroll
    for (int bt = 0; bt < 4; ++bt)
#pragma unroll
      for (int v = 0; v < 8; ++v) {
        const int b = bt * 16 + v + 8 * half;
        const float hv = hval[bt][v];
        sh[shswz(b, j)] = f2bf(hv);                                   // h(t) -> LDS
        hs[((size_t)b * Tn + tt) * (2 * Hn) + colOff + j] = hv;       // h(t) -> global
      }
    __syncthreads();  // h(t) visible for next step
  }
}

// ---------------------------------------------------------------- CRF Viterbi
// One wave per batch row, lane = tag class.
__global__ __launch_bounds__(1024) void crf_viterbi(
    const float* __restrict__ em,            // (B,T,C)
    const unsigned char* __restrict__ mask,  // (B,T)
    const float* __restrict__ start_trans,
    const float* __restrict__ end_trans,
    const float* __restrict__ trans,         // (C,C)
    unsigned char* __restrict__ hist,        // (T-1,B,C)
    float* __restrict__ out_tags,            // (B,T) as float
    float* __restrict__ out_score) {         // (B,)
  const int lane = threadIdx.x & 31;
  const int b = blockIdx.x * 32 + (threadIdx.x >> 5);

  float tc[Cn];  // lane j caches trans[:, j]
#pragma unroll
  for (int i = 0; i < Cn; ++i) tc[i] = trans[i * Cn + lane];

  float score = start_trans[lane] + em[(size_t)b * Tn * Cn + lane];

  for (int t = 1; t < Tn; ++t) {
    float best = -1e30f; int arg = 0;
#pragma unroll
    for (int i = 0; i < Cn; ++i) {
      float cand = __shfl(score, i, 32) + tc[i];
      if (cand > best) { best = cand; arg = i; }
    }
    bool m = mask[b * Tn + t] != 0;
    float ns = best + em[((size_t)b * Tn + t) * Cn + lane];
    score = m ? ns : score;
    arg   = m ? arg : lane;
    hist[((size_t)(t - 1) * Bn + b) * Cn + lane] = (unsigned char)arg;
  }

  float fin = score + end_trans[lane];
  float bs = fin; int btag = lane;
#pragma unroll
  for (int off = 16; off; off >>= 1) {       // argmax, first-index tie-break
    float os = __shfl_xor(bs, off, 32);
    int   ot = __shfl_xor(btag, off, 32);
    if (os > bs || (os == bs && ot < btag)) { bs = os; btag = ot; }
  }

  if (lane == 0) {
    out_score[b] = bs;
    int tag = btag;
    out_tags[(size_t)b * Tn + (Tn - 1)] = (float)tag;
    for (int t = Tn - 2; t >= 0; --t) {
      tag = hist[((size_t)t * Bn + b) * Cn + tag];
      out_tags[(size_t)b * Tn + t] = (float)tag;
    }
  }
}

// ---------------------------------------------------------------- host
extern "C" void kernel_launch(void* const* d_in, const int* in_sizes, int n_in,
                              void* d_out, int out_size, void* d_ws, size_t ws_size,
                              hipStream_t stream) {
  const float* x      = (const float*)d_in[0];
  const unsigned char* mask = (const unsigned char*)d_in[1];
  const float* W_ih0  = (const float*)d_in[2];
  const float* W_hh0  = (const float*)d_in[3];
  const float* b0     = (const float*)d_in[4];
  const float* W_ih1  = (const float*)d_in[5];
  const float* W_hh1  = (const float*)d_in[6];
  const float* b1     = (const float*)d_in[7];
  const float* W_out  = (const float*)d_in[8];
  const float* b_out  = (const float*)d_in[9];
  const float* start_trans = (const float*)d_in[10];
  const float* end_trans   = (const float*)d_in[11];
  const float* trans       = (const float*)d_in[12];

  char* ws = (char*)d_ws;
  size_t off = 0;
  auto alloc = [&](size_t bytes) -> char* {
    char* p = ws + off;
    off += (bytes + 255) & ~(size_t)255;
    return p;
  };

  const int Mbt = Bn * Tn;  // 16384 rows for the big GEMMs
  ushort_t* xb    = (ushort_t*)alloc((size_t)Mbt * Fn * 2);
  ushort_t* wih0b = (ushort_t*)alloc((size_t)2 * 4 * Hn * Fn * 2);
  ushort_t* whh0b = (ushort_t*)alloc((size_t)2 * 4 * Hn * Hn * 2);
  ushort_t* wih1b = (ushort_t*)alloc((size_t)2 * 4 * Hn * 2 * Hn * 2);
  ushort_t* whh1b = (ushort_t*)alloc((size_t)2 * 4 * Hn * Hn * 2);
  ushort_t* woutb = (ushort_t*)alloc((size_t)Cn * 2 * Hn * 2);
  float*    preb  = (float*)alloc((size_t)2 * Mbt * 4 * Hn * 4);  // reused both layers
  float*    hf    = (float*)alloc((size_t)Mbt * 2 * Hn * 4);      // layer out (reused)
  ushort_t* hb    = (ushort_t*)alloc((size_t)Mbt * 2 * Hn * 2);   // layer out bf16
  float*    emis  = (float*)alloc((size_t)Mbt * Cn * 4);
  unsigned char* hist = (unsigned char*)alloc((size_t)(Tn - 1) * Bn * Cn);

  auto cvt = [&](const float* in, ushort_t* outp, long long n) {
    cvt_bf16<<<(unsigned)((n + 1023) / 1024), 1024, 0, stream>>>(in, outp, n);
  };
  auto gemm_big = [&](const ushort_t* A, int lda, const ushort_t* W, int ldw,
                      const float* bias, float* C, int ldc, int M, int N, int K) {
    gemm_bf16_lds<<<(M / BM) * (N / BN), 256, 0, stream>>>(A, lda, W, ldw, bias,
                                                           C, ldc, M, N, K);
  };

  // Stage weights/activations to bf16 (operands); accumulation stays f32.
  cvt(x,     xb,    (long long)Mbt * Fn);
  cvt(W_ih0, wih0b, 2LL * 4 * Hn * Fn);
  cvt(W_hh0, whh0b, 2LL * 4 * Hn * Hn);
  cvt(W_ih1, wih1b, 2LL * 4 * Hn * 2 * Hn);
  cvt(W_hh1, whh1b, 2LL * 4 * Hn * Hn);
  cvt(W_out, woutb, (long long)Cn * 2 * Hn);

  // Layer 0: input projection (both dirs), then fwd+bwd recurrence in one launch
  for (int d = 0; d < 2; ++d)
    gemm_big(xb, Fn, wih0b + (size_t)d * 4 * Hn * Fn, Fn, b0 + (size_t)d * 4 * Hn,
             preb + (size_t)d * Mbt * 4 * Hn, 4 * Hn, Mbt, 4 * Hn, Fn);
  bilstm_rec<<<2, 1024, 0, stream>>>(preb, preb + (size_t)Mbt * 4 * Hn,
                                     whh0b, whh0b + (size_t)4 * Hn * Hn, hf);
  cvt(hf, hb, (long long)Mbt * 2 * Hn);

  // Layer 1
  for (int d = 0; d < 2; ++d)
    gemm_big(hb, 2 * Hn, wih1b + (size_t)d * 4 * Hn * 2 * Hn, 2 * Hn,
             b1 + (size_t)d * 4 * Hn,
             preb + (size_t)d * Mbt * 4 * Hn, 4 * Hn, Mbt, 4 * Hn, 2 * Hn);
  bilstm_rec<<<2, 1024, 0, stream>>>(preb, preb + (size_t)Mbt * 4 * Hn,
                                     whh1b, whh1b + (size_t)4 * Hn * Hn, hf);
  cvt(hf, hb, (long long)Mbt * 2 * Hn);

  // Emissions (N=32 -> direct kernel) + Viterbi
  {
    int tiles = (Mbt / 16) * (Cn / 16);
    gemm_bf16_wmma<<<tiles / 8, 256, 0, stream>>>(hb, 2 * Hn, woutb, 2 * Hn, b_out,
                                                  emis, Cn, Mbt, Cn, 2 * Hn);
  }
  float* out = (float*)d_out;
  crf_viterbi<<<2, 1024, 0, stream>>>(emis, mask, start_trans, end_trans, trans,
                                      hist, out, out + (size_t)Bn * Tn);

  (void)in_sizes; (void)n_in; (void)out_size; (void)ws_size;
}
